// GraphormerMultiHeadAttention_76914274337210
// MI455X (gfx1250) — matrix operand order repaired and checked by reading, not compiled, not run on hardware
//
#include <hip/hip_runtime.h>
#include <math.h>

typedef __attribute__((ext_vector_type(16))) _Float16 v16h;
typedef __attribute__((ext_vector_type(8)))  _Float16 v8h;
typedef __attribute__((ext_vector_type(2)))  _Float16 h2;
typedef __attribute__((ext_vector_type(8)))  float    v8f;
typedef __attribute__((ext_vector_type(4)))  unsigned int u32x4;
typedef __attribute__((ext_vector_type(4)))  int          i32x4;
typedef __attribute__((ext_vector_type(8)))  int          i32x8;

#define N_NODES 4096
#define DMODEL  256
#define NHEAD   8
#define HDIM    32
#define NEDGE   65536
#define JB      256   // K/V chunk staged in LDS per workgroup
#define NWAVE   8     // waves per flash-attention workgroup

// ---------------------------------------------------------------------------
// WMMA fragment loaders (CDNA5 wave32 layouts, CDNA5 ISA §7.12.2)
// A 16x32 f16: lane row = lane&15; kb=(lane>>4)*8; a[0..7]=K kb..kb+7,
//              a[8..15]=K kb+16..kb+23  -> two 16B contiguous chunks.
// B 32x16 f16: lane col = lane&15 (column baked into base pointer);
//              ko=(lane>>4)*8; b[0..7]=K ko..ko+7, b[8..15]=K ko+16..ko+23.
// Work for both global (global_load_b128) and LDS (ds_read_b128) pointers.
// ---------------------------------------------------------------------------
__device__ __forceinline__ v16h load_a_16x32(const _Float16* base, int row_stride, int lane) {
    int r  = lane & 15;
    int kb = (lane >> 4) * 8;
    const _Float16* p = base + r * row_stride + kb;
    v8h lo = *(const v8h*)(p);
    v8h hi = *(const v8h*)(p + 16);
    v16h a;
#pragma unroll
    for (int i = 0; i < 8; ++i) { a[i] = lo[i]; a[i + 8] = hi[i]; }
    return a;
}

__device__ __forceinline__ v16h load_b_32x16(const _Float16* col_base, int lane) {
    int ko = (lane >> 4) * 8;
    v8h lo = *(const v8h*)(col_base + ko);
    v8h hi = *(const v8h*)(col_base + ko + 16);
    v16h b;
#pragma unroll
    for (int i = 0; i < 8; ++i) { b[i] = lo[i]; b[i + 8] = hi[i]; }
    return b;
}

#define WMMA_F16(A, B, C) \
    __builtin_amdgcn_wmma_f32_16x16x32_f16(false, (A), false, (B), (short)0, (C), false, false)

// ---------------------------------------------------------------------------
// Tensor Data Mover: 2-D tile (tile_w x tile_h elements, 2-byte data) from
// global (row stride `stride` elements) into contiguous LDS at lds_off.
// D# layout per CDNA5 ISA. Issued by one wave;
// EXEC is ignored by TENSOR ops. Tracked by TENSORcnt.
// ---------------------------------------------------------------------------
__device__ __forceinline__ void tdm_load_2d_f16(const void* gptr, unsigned int lds_off,
                                                unsigned int tile_w, unsigned int tile_h,
                                                unsigned int stride_elems) {
    unsigned long long ga = (unsigned long long)(uintptr_t)gptr;
    u32x4 g0;
    g0[0] = 1u;                                            // count=1, user mode
    g0[1] = lds_off;                                       // LDS byte address
    g0[2] = (unsigned int)(ga & 0xFFFFFFFFull);            // global_addr[31:0]
    g0[3] = (unsigned int)((ga >> 32) & 0x1FFFFFFull)      // global_addr[56:32]
          | (2u << 30);                                    // type = 2 (image)
    i32x8 g1;
    g1[0] = (1 << 16);                                     // data_size=1 -> 2 bytes
    g1[1] = (int)((tile_w & 0xFFFFu) << 16);               // tensor_dim0 lo16
    g1[2] = (int)(((tile_w >> 16) & 0xFFFFu)               // tensor_dim0 hi16
          | ((tile_h & 0xFFFFu) << 16));                   // tensor_dim1 lo16
    g1[3] = (int)(((tile_h >> 16) & 0xFFFFu)               // tensor_dim1 hi16
          | ((tile_w & 0xFFFFu) << 16));                   // tile_dim0
    g1[4] = (int)(tile_h & 0xFFFFu);                       // tile_dim1 (tile_dim2=0)
    g1[5] = (int)stride_elems;                             // tensor_dim0_stride lo32
    g1[6] = 0;                                             // stride hi / dim1_stride lo
    g1[7] = 0;
    i32x4 z4 = {0, 0, 0, 0};
#if defined(__clang_major__) && (__clang_major__ >= 23)
    i32x8 z8 = {0, 0, 0, 0, 0, 0, 0, 0};
    __builtin_amdgcn_tensor_load_to_lds(g0, g1, z4, z4, z8, 0);
#else
    __builtin_amdgcn_tensor_load_to_lds(g0, g1, z4, z4, 0);
#endif
}

// ---------------------------------------------------------------------------
// Packing kernels
// ---------------------------------------------------------------------------
__global__ void pack_f32_to_f16(const float* __restrict__ src, _Float16* __restrict__ dst, int n) {
    int i = blockIdx.x * blockDim.x + threadIdx.x;
    if (i < n) dst[i] = (_Float16)src[i];
}

// Wt[n][k] = W[k][n], so GEMM B fragments are contiguous along K.
__global__ void pack_w_transpose(const float* __restrict__ W, _Float16* __restrict__ Wt) {
    int idx = blockIdx.x * blockDim.x + threadIdx.x;
    int n = idx & 255, k = idx >> 8;
    Wt[n * DMODEL + k] = (_Float16)W[k * DMODEL + n];
}

__global__ void zero_f32(float* __restrict__ p, int n) {
    int i = blockIdx.x * blockDim.x + threadIdx.x;
    if (i < n) p[i] = 0.0f;
}

// col_bias[h][tgt[e]] = (edge_attr[e] @ We + be)[h]  (last-write-wins race ok)
__global__ void edge_bias_scatter(const int* __restrict__ edge_index,
                                  const float* __restrict__ edge_attr,
                                  const float* __restrict__ We,
                                  const float* __restrict__ be,
                                  float* __restrict__ col_bias) {
    int e = blockIdx.x * blockDim.x + threadIdx.x;
    if (e >= NEDGE) return;
    int tgt = edge_index[NEDGE + e];
    float a0 = edge_attr[e * 3 + 0];
    float a1 = edge_attr[e * 3 + 1];
    float a2 = edge_attr[e * 3 + 2];
#pragma unroll
    for (int h = 0; h < NHEAD; ++h) {
        float v = a0 * We[0 * NHEAD + h] + a1 * We[1 * NHEAD + h] + a2 * We[2 * NHEAD + h] + be[h];
        col_bias[h * N_NODES + tgt] = v;
    }
}

// ---------------------------------------------------------------------------
// WMMA GEMM: C[4096x256] = A_f16 @ Bt_f16 + bias. 32x32 tile per wave
// (4 accumulators, A/B fragment reuse -> 4 WMMAs per 2 A + 2 B loads).
// ---------------------------------------------------------------------------
#define MODE_F16_RM 0
#define MODE_F16_VT 1
#define MODE_F32_RM 2

template <int MODE>
__device__ __forceinline__ void store_tile(v8f acc, int m0, int n0, int ln, int hi,
                                           const float* __restrict__ bias,
                                           _Float16* __restrict__ out16,
                                           float* __restrict__ out32, float scale) {
    int col = n0 + ln;
    float bv = bias[col];
#pragma unroll
    for (int r = 0; r < 8; ++r) {
        int row = m0 + r + hi * 8;
        float val = (acc[r] + bv) * scale;
        if constexpr (MODE == MODE_F16_RM) {
            out16[row * DMODEL + col] = (_Float16)val;
        } else if constexpr (MODE == MODE_F16_VT) {
            out16[col * N_NODES + row] = (_Float16)val;   // vt[h*32+d][row]
        } else {
            out32[row * DMODEL + col] = val;
        }
    }
}

template <int MODE>
__global__ __launch_bounds__(32)
void wmma_gemm(const _Float16* __restrict__ A,   // [4096][256] row-major
               const _Float16* __restrict__ Bt,  // [256][256]  Bt[n][k]
               const float* __restrict__ bias,   // [256]
               _Float16* __restrict__ out16,
               float* __restrict__ out32,
               float scale) {
    int lane = threadIdx.x & 31;
    int m0 = blockIdx.x * 32;
    int n0 = blockIdx.y * 32;
    int ln = lane & 15;
    int hi = (lane >> 4) & 1;

    v8f acc00 = {}, acc01 = {}, acc10 = {}, acc11 = {};
    const _Float16* bc0 = Bt + (n0 + ln) * DMODEL;
    const _Float16* bc1 = Bt + (n0 + 16 + ln) * DMODEL;
#pragma unroll 2
    for (int k0 = 0; k0 < DMODEL; k0 += 32) {
        v16h a0 = load_a_16x32(A + m0 * DMODEL + k0, DMODEL, lane);
        v16h a1 = load_a_16x32(A + (m0 + 16) * DMODEL + k0, DMODEL, lane);
        v16h b0 = load_b_32x16(bc0 + k0, lane);
        v16h b1 = load_b_32x16(bc1 + k0, lane);
        acc00 = WMMA_F16(a0, b0, acc00);
        acc01 = WMMA_F16(a0, b1, acc01);
        acc10 = WMMA_F16(a1, b0, acc10);
        acc11 = WMMA_F16(a1, b1, acc11);
    }
    store_tile<MODE>(acc00, m0,      n0,      ln, hi, bias, out16, out32, scale);
    store_tile<MODE>(acc01, m0,      n0 + 16, ln, hi, bias, out16, out32, scale);
    store_tile<MODE>(acc10, m0 + 16, n0,      ln, hi, bias, out16, out32, scale);
    store_tile<MODE>(acc11, m0 + 16, n0 + 16, ln, hi, bias, out16, out32, scale);
}

// ---------------------------------------------------------------------------
// Flash attention: 8 waves / workgroup, each owns a 16-row block of one head.
// K/V chunks (JB=256 keys) staged into LDS once per workgroup by the TDM,
// then all waves compute from LDS. Per 32-key step:
//   2 score WMMAs, packed-f16 row-max butterfly (16 bpermutes), exps,
//   P C->A relayout through per-wave LDS, 3 AV-side WMMAs (o_lo, o_hi, and
//   the softmax denominator accumulated as P @ ones on the matrix pipe).
// ---------------------------------------------------------------------------
__global__ __launch_bounds__(NWAVE * 32)
void flash_attn(const _Float16* __restrict__ qh,   // [4096][256], pre-scaled 1/sqrt(hd)
                const _Float16* __restrict__ kh,   // [4096][256]
                const _Float16* __restrict__ vt,   // [256][4096] per-head transposed
                const float* __restrict__ col_bias,// [8][4096]
                _Float16* __restrict__ oh) {       // [4096][256]
    __shared__ _Float16 k_tile[JB * HDIM];          // [j][d]  16 KB
    __shared__ _Float16 v_tile[HDIM * JB];          // [d][j]  16 KB
    __shared__ _Float16 p_lds[NWAVE * 16 * 32];     // per-wave P staging, 8 KB

    int tid  = threadIdx.x;
    int wave = tid >> 5;
    int lane = tid & 31;
    int ln = lane & 15;
    int hi = (lane >> 4) & 1;
    int h  = blockIdx.y;
    int i0 = (blockIdx.x * NWAVE + wave) * 16;

    v16h qa = load_a_16x32(qh + i0 * DMODEL + h * HDIM, DMODEL, lane);

    v16h ones_b;
#pragma unroll
    for (int i = 0; i < 16; ++i) ones_b[i] = (_Float16)1.0f;

    v8f o0 = {}, o1 = {}, lacc = {};
    float mrow[8];
#pragma unroll
    for (int r = 0; r < 8; ++r) mrow[r] = -1e30f;

    const float* biasrow = col_bias + h * N_NODES;
    _Float16* pw = p_lds + wave * (16 * 32);
    unsigned int k_off = (unsigned int)(uintptr_t)(&k_tile[0]);  // LDS byte offset
    unsigned int v_off = (unsigned int)(uintptr_t)(&v_tile[0]);

    for (int jc = 0; jc < N_NODES; jc += JB) {
        __syncthreads();   // previous chunk fully consumed before overwrite
        if (tid < 32) {    // wave 0 drives the TDM
            tdm_load_2d_f16(kh + jc * DMODEL + h * HDIM, k_off, HDIM, JB, DMODEL);
            tdm_load_2d_f16(vt + (h * HDIM) * N_NODES + jc, v_off, JB, HDIM, N_NODES);
            __builtin_amdgcn_s_wait_tensorcnt(0);
        }
        __syncthreads();   // staged K/V visible to all waves

        for (int jl = 0; jl < JB; jl += 32) {
            // ---- scores: S = Q . K^T  (B frag from LDS: col j, K over d)
            v16h kb0 = load_b_32x16(k_tile + (jl + ln) * HDIM, lane);
            v16h kb1 = load_b_32x16(k_tile + (jl + 16 + ln) * HDIM, lane);
            v8f zz = {};
            v8f s0 = WMMA_F16(qa, kb0, zz);
            v8f s1 = WMMA_F16(qa, kb1, zz);

            float b0 = biasrow[jc + jl + ln];
            float b1 = biasrow[jc + jl + 16 + ln];
#pragma unroll
            for (int r = 0; r < 8; ++r) { s0[r] += b0; s1[r] += b1; }

            // ---- packed-f16 row-max butterfly across the 16-lane group
            h2 pk[4];
#pragma unroll
            for (int rr = 0; rr < 4; ++rr) {
                h2 c;
                c.x = (_Float16)fmaxf(s0[2 * rr],     s1[2 * rr]);
                c.y = (_Float16)fmaxf(s0[2 * rr + 1], s1[2 * rr + 1]);
                pk[rr] = c;
            }
#pragma unroll
            for (int mask = 1; mask < 16; mask <<= 1) {
#pragma unroll
                for (int rr = 0; rr < 4; ++rr) {
                    int o = __shfl_xor(__builtin_bit_cast(int, pk[rr]), mask, 32);
                    pk[rr] = __builtin_elementwise_max(pk[rr], __builtin_bit_cast(h2, o));
                }
            }

            // ---- online rescale + exponentials; P tile to per-wave LDS
#pragma unroll
            for (int r = 0; r < 8; ++r) {
                float cmax = (float)((r & 1) ? pk[r >> 1].y : pk[r >> 1].x);
                float mnew = fmaxf(mrow[r], cmax);
                float corr = __expf(mrow[r] - mnew);
                float p0 = __expf(s0[r] - mnew);
                float p1 = __expf(s1[r] - mnew);
                mrow[r] = mnew;
                o0[r]   *= corr;
                o1[r]   *= corr;
                lacc[r] *= corr;
                pw[(r + hi * 8) * 32 + ln]      = (_Float16)p0;
                pw[(r + hi * 8) * 32 + 16 + ln] = (_Float16)p1;
            }
            asm volatile("s_wait_dscnt 0x0" ::: "memory");

            // ---- P: C-layout -> A-layout (16x32)
            v16h pa;
            {
                const _Float16* pr = pw + ln * 32 + hi * 8;
                v8h lo = *(const v8h*)(pr);
                v8h hh = *(const v8h*)(pr + 16);
#pragma unroll
                for (int i = 0; i < 8; ++i) { pa[i] = lo[i]; pa[i + 8] = hh[i]; }
            }

            // ---- O += P.V ; denominator += P.1 (all on the matrix pipe)
            v16h vb0 = load_b_32x16(v_tile + ln * JB + jl, lane);
            v16h vb1 = load_b_32x16(v_tile + (16 + ln) * JB + jl, lane);
            o0   = WMMA_F16(pa, vb0, o0);
            o1   = WMMA_F16(pa, vb1, o1);
            lacc = WMMA_F16(pa, ones_b, lacc);
        }
    }

    // ---- normalize + store (f16 row-major, feeds output projection GEMM)
#pragma unroll
    for (int r = 0; r < 8; ++r) {
        int row = i0 + r + hi * 8;
        float inv = 1.0f / lacc[r];
        oh[row * DMODEL + h * HDIM + ln]      = (_Float16)(o0[r] * inv);
        oh[row * DMODEL + h * HDIM + 16 + ln] = (_Float16)(o1[r] * inv);
    }
}

// ---------------------------------------------------------------------------
// Host-side orchestration
// ---------------------------------------------------------------------------
extern "C" void kernel_launch(void* const* d_in, const int* in_sizes, int n_in,
                              void* d_out, int out_size, void* d_ws, size_t ws_size,
                              hipStream_t stream) {
    const float* x          = (const float*)d_in[0];
    const int*   edge_index = (const int*)  d_in[1];
    const float* edge_attr  = (const float*)d_in[2];
    const float* Wq = (const float*)d_in[3];
    const float* bq = (const float*)d_in[4];
    const float* Wk = (const float*)d_in[5];
    const float* bk = (const float*)d_in[6];
    const float* Wv = (const float*)d_in[7];
    const float* bv = (const float*)d_in[8];
    const float* Wo = (const float*)d_in[9];
    const float* bo = (const float*)d_in[10];
    const float* We = (const float*)d_in[11];
    const float* be = (const float*)d_in[12];
    float* out = (float*)d_out;

    // Workspace carve-up (256B aligned); total ~10.6 MB.
    char* ws = (char*)d_ws;
    auto carve = [&](size_t bytes) -> char* {
        char* p = ws;
        ws += (bytes + 255) & ~(size_t)255;
        return p;
    };
    _Float16* xh  = (_Float16*)carve((size_t)N_NODES * DMODEL * 2);
    _Float16* qh  = (_Float16*)carve((size_t)N_NODES * DMODEL * 2);
    _Float16* kh  = (_Float16*)carve((size_t)N_NODES * DMODEL * 2);
    _Float16* vt  = (_Float16*)carve((size_t)DMODEL * N_NODES * 2);
    _Float16* ah  = (_Float16*)carve((size_t)N_NODES * DMODEL * 2);
    _Float16* Wtq = (_Float16*)carve((size_t)DMODEL * DMODEL * 2);
    _Float16* Wtk = (_Float16*)carve((size_t)DMODEL * DMODEL * 2);
    _Float16* Wtv = (_Float16*)carve((size_t)DMODEL * DMODEL * 2);
    _Float16* Wto = (_Float16*)carve((size_t)DMODEL * DMODEL * 2);
    float*    cb  = (float*)   carve((size_t)NHEAD * N_NODES * 4);

    const int TPB = 256;

    // Pack x and weights to f16 (weights transposed for contiguous B frags)
    pack_f32_to_f16<<<(N_NODES * DMODEL + TPB - 1) / TPB, TPB, 0, stream>>>(x, xh, N_NODES * DMODEL);
    pack_w_transpose<<<(DMODEL * DMODEL) / TPB, TPB, 0, stream>>>(Wq, Wtq);
    pack_w_transpose<<<(DMODEL * DMODEL) / TPB, TPB, 0, stream>>>(Wk, Wtk);
    pack_w_transpose<<<(DMODEL * DMODEL) / TPB, TPB, 0, stream>>>(Wv, Wtv);
    pack_w_transpose<<<(DMODEL * DMODEL) / TPB, TPB, 0, stream>>>(Wo, Wto);

    // Edge column bias
    zero_f32<<<(NHEAD * N_NODES) / TPB, TPB, 0, stream>>>(cb, NHEAD * N_NODES);
    edge_bias_scatter<<<NEDGE / TPB, TPB, 0, stream>>>(edge_index, edge_attr, We, be, cb);

    // QKV projections (q pre-scaled by 1/sqrt(hd); v stored per-head transposed)
    dim3 ggrid(N_NODES / 32, DMODEL / 32);
    const float qscale = 0.17677669529663687f;  // 1/sqrt(32)
    wmma_gemm<MODE_F16_RM><<<ggrid, 32, 0, stream>>>(xh, Wtq, bq, qh, nullptr, qscale);
    wmma_gemm<MODE_F16_RM><<<ggrid, 32, 0, stream>>>(xh, Wtk, bk, kh, nullptr, 1.0f);
    wmma_gemm<MODE_F16_VT><<<ggrid, 32, 0, stream>>>(xh, Wtv, bv, vt, nullptr, 1.0f);

    // Flash attention (8 waves/WG, TDM-staged K/V chunks in LDS)
    dim3 agrid(N_NODES / (16 * NWAVE), NHEAD);
    flash_attn<<<agrid, NWAVE * 32, 0, stream>>>(qh, kh, vt, cb, ah);

    // Output projection (f32 to d_out)
    wmma_gemm<MODE_F32_RM><<<ggrid, 32, 0, stream>>>(ah, Wto, bo, nullptr, out, 1.0f);

    (void)in_sizes; (void)n_in; (void)out_size; (void)ws_size;
}
